// LagrangeInterp_RNN_18897856103014
// MI455X (gfx1250) — compile-verified, matrix operand order checked by reading, not run on hardware
//
#include <hip/hip_runtime.h>
#include <math.h>

#define B_DIM 32
#define T_DIM 1024
#define I_DIM 512
#define H_DIM 512

typedef __attribute__((ext_vector_type(16))) __bf16         bf16x16;
typedef __attribute__((ext_vector_type(8)))  float          f32x8;
typedef __attribute__((ext_vector_type(4)))  float          f32x4;
typedef __attribute__((ext_vector_type(4)))  unsigned int   u32x4;
typedef __attribute__((ext_vector_type(8)))  unsigned short u16x8;
typedef __attribute__((ext_vector_type(16))) unsigned short u16x16;

// Native bf16 converts: backend emits v_cvt_pk_bf16_f32.
static __device__ __forceinline__ unsigned short f2bf(float f) {
    __bf16 h = (__bf16)f;
    return __builtin_bit_cast(unsigned short, h);
}
static __device__ __forceinline__ float bf2f(unsigned short u) {
    return (float)__builtin_bit_cast(__bf16, u);
}
static __device__ __forceinline__ u16x8 cvt8(f32x4 a, f32x4 b) {
    u16x8 o;
    o[0] = f2bf(a[0]); o[1] = f2bf(a[1]); o[2] = f2bf(a[2]); o[3] = f2bf(a[3]);
    o[4] = f2bf(b[0]); o[5] = f2bf(b[1]); o[6] = f2bf(b[2]); o[7] = f2bf(b[3]);
    return o;
}
static __device__ __forceinline__ u16x16 cat16(u16x8 lo, u16x8 hi) {
    return __builtin_shufflevector(lo, hi, 0,1,2,3,4,5,6,7,8,9,10,11,12,13,14,15);
}
static __device__ __forceinline__ f32x8 wmma_bf16(u16x16 a, u16x16 b, f32x8 c) {
    return __builtin_amdgcn_wmma_f32_16x16x32_bf16(
        false, __builtin_bit_cast(bf16x16, a),
        false, __builtin_bit_cast(bf16x16, b),
        (short)0, c, false, false);
}

// Hardware tanh (gfx1250 V_TANH_F32) with a cheap exp2-based fallback.
static __device__ __forceinline__ float fast_tanh(float x) {
#if __has_builtin(__builtin_amdgcn_tanhf)
    return __builtin_amdgcn_tanhf(x);
#else
    // tanh(x) = 1 - 2/(exp(2x)+1); robust at +/-inf (-> +/-1).
    float e = __builtin_amdgcn_exp2f(x * 2.8853900817779268f); // 2*log2(e)
    return 1.0f - 2.0f * __builtin_amdgcn_rcpf(e + 1.0f);
#endif
}

// ---------------------------------------------------------------------------
// Kernel 0: convert Wih / Whh from f32 to bf16 in workspace
// ---------------------------------------------------------------------------
__global__ __launch_bounds__(256) void cvt_weights(
    const float* __restrict__ Wih, const float* __restrict__ Whh,
    unsigned short* __restrict__ WihB, unsigned short* __restrict__ WhhB) {
    int i = blockIdx.x * blockDim.x + threadIdx.x;
    if (i < H_DIM * I_DIM) {
        WihB[i] = f2bf(Wih[i]);
        WhhB[i] = f2bf(Whh[i]);
    }
}

// ---------------------------------------------------------------------------
// Kernel 1: xproj[m, n] = sum_k x[m, k] * Wih[n, k]   (bf16 WMMA, f32 acc)
//   m = b*T + t (row-major x [B,T,I]); output stored bf16 (biases added later)
//   Block: 256 threads = 8 waves arranged 2 (M) x 4 (N); wave tile 32x32.
//   Workgroup tile: 64 (M) x 128 (N).  Grid: (32768/64, 512/128) = (512, 4)
// ---------------------------------------------------------------------------
__global__ __launch_bounds__(256) void xproj_gemm(
    const float* __restrict__ x, const unsigned short* __restrict__ WihB,
    unsigned short* __restrict__ xproj) {
    const int l  = threadIdx.x & 31;
    const int w  = threadIdx.x >> 5;
    const int lr = l & 15;
    const int hi = l >> 4;                 // 0 / 1: which K half this lane owns
    const int mOff = blockIdx.x * 64 + (w & 1) * 32;
    const int nOff = blockIdx.y * 128 + (w >> 1) * 32;

    f32x8 acc[2][2] = {};

    for (int kk = 0; kk < 16; ++kk) {      // K chunks of 32
        const int kb = kk * 32 + hi * 8;
        u16x16 af[2], bf[2];
        #pragma unroll
        for (int mt = 0; mt < 2; ++mt) {
            const float* ap = x + (size_t)(mOff + mt * 16 + lr) * I_DIM + kb;
            f32x4 c0 = *(const f32x4*)(ap);
            f32x4 c1 = *(const f32x4*)(ap + 4);
            f32x4 c2 = *(const f32x4*)(ap + 16);
            f32x4 c3 = *(const f32x4*)(ap + 20);
            af[mt] = cat16(cvt8(c0, c1), cvt8(c2, c3));
        }
        #pragma unroll
        for (int nt = 0; nt < 2; ++nt) {
            const unsigned short* bp =
                WihB + (size_t)(nOff + nt * 16 + lr) * I_DIM + kb;
            u16x8 b0 = __builtin_bit_cast(u16x8, *(const u32x4*)(bp));
            u16x8 b1 = __builtin_bit_cast(u16x8, *(const u32x4*)(bp + 16));
            bf[nt] = cat16(b0, b1);
        }
        #pragma unroll
        for (int mt = 0; mt < 2; ++mt)
            #pragma unroll
            for (int nt = 0; nt < 2; ++nt)
                acc[mt][nt] = wmma_bf16(af[mt], bf[nt], acc[mt][nt]);
    }

    #pragma unroll
    for (int mt = 0; mt < 2; ++mt)
        #pragma unroll
        for (int nt = 0; nt < 2; ++nt) {
            const int n = nOff + nt * 16 + lr;
            #pragma unroll
            for (int r = 0; r < 8; ++r) {
                const int m = mOff + mt * 16 + (hi ? r + 8 : r);
                xproj[(size_t)m * H_DIM + n] = f2bf(acc[mt][nt][r]);
            }
        }
}

// ---------------------------------------------------------------------------
// Kernel 2: sequential Lagrange-RNN scan.
//   Batches are independent in the recurrence -> TWO workgroups, each owning
//   16 batches (one 16-row WMMA M-tile), on separate WGPs in parallel
//   (the LDS request is padded >160 KB so two WGs cannot share a WGP).
//   1024 threads = 32 waves; wave w owns output columns [16w, 16w+16).
//   LDS: 4-slot bf16 ring (64 KB) + h_read staging (16.3 KB).
//   Whh bf16 slice per wave: 16 register-resident B-fragments.
// ---------------------------------------------------------------------------
#define MB 16                           // batches per workgroup
#define SLOT_STRIDE (MB * H_DIM)        // 8192 bf16 elements per ring slot
#define HREAD_PITCH (H_DIM + 8)         // padded pitch (bank-conflict free)

__global__ __launch_bounds__(1024) void rnn_scan(
    const unsigned short* __restrict__ WhhB,
    const unsigned short* __restrict__ xproj,
    const float* __restrict__ bih, const float* __restrict__ bhh,
    const float* __restrict__ taps,
    float* __restrict__ out_states, float* __restrict__ out_last) {
    extern __shared__ unsigned short smem[];
    unsigned short* ring  = smem;                      // 4 * 8192 bf16
    unsigned short* hread = smem + 4 * SLOT_STRIDE;    // 16 * 520 bf16

    const int tid  = threadIdx.x;
    const int l    = tid & 31;
    const int w    = tid >> 5;       // wave id 0..31 -> N tile
    const int lr   = l & 15;
    const int hi   = l >> 4;
    const int n    = w * 16 + lr;    // output column owned by this lane
    const int bOff = blockIdx.x * MB;

    // Preload this wave's 512x16 Whh slice as 16 B-fragments (registers).
    u16x16 bw[16];
    #pragma unroll
    for (int kf = 0; kf < 16; ++kf) {
        const unsigned short* bp = WhhB + (size_t)n * H_DIM + kf * 32 + hi * 8;
        u16x8 b0 = __builtin_bit_cast(u16x8, *(const u32x4*)(bp));
        u16x8 b1 = __builtin_bit_cast(u16x8, *(const u32x4*)(bp + 16));
        bw[kf] = cat16(b0, b1);
    }
    const float bsum = bih[n] + bhh[n];
    const float k0 = taps[0], k1 = taps[1], k2 = taps[2], k3 = taps[3];

    // Zero the ring (initial history = 0).
    for (int i = tid; i < 4 * SLOT_STRIDE / 8; i += 1024)
        *(u16x8*)(ring + i * 8) = (u16x8)0;
    __syncthreads();

    // h_read mapping: exactly one 8-element chunk per thread.
    const int hm = tid >> 6;         // row 0..15 (batch within tile)
    const int hk = (tid & 63) * 8;   // k base

    // xproj C-fragment loader (software-pipelined one step ahead).
    auto load_xp = [&](int t) {
        f32x8 c;
        #pragma unroll
        for (int r = 0; r < 8; ++r) {
            const int m = bOff + (hi ? r + 8 : r);
            c[r] = bf2f(xproj[((size_t)m * T_DIM + t) * H_DIM + n]);
        }
        return c;
    };
    f32x8 xp = load_xp(0);

    for (int t = 0; t < T_DIM; ++t) {
        const int s0 = (t + 3) & 3, s1 = (t + 2) & 3,
                  s2 = (t + 1) & 3, s3 = t & 3;

        // ---- Phase A: h_read = sum_j tap[j] * ring[t-1-j] (f32 combine) ----
        {
            const int off = hm * H_DIM + hk;
            u16x8 a0 = *(const u16x8*)(ring + s0 * SLOT_STRIDE + off);
            u16x8 a1 = *(const u16x8*)(ring + s1 * SLOT_STRIDE + off);
            u16x8 a2 = *(const u16x8*)(ring + s2 * SLOT_STRIDE + off);
            u16x8 a3 = *(const u16x8*)(ring + s3 * SLOT_STRIDE + off);
            u16x8 o;
            #pragma unroll
            for (int i = 0; i < 8; ++i) {
                float v = k0 * bf2f(a0[i]) + k1 * bf2f(a1[i]) +
                          k2 * bf2f(a2[i]) + k3 * bf2f(a3[i]);
                o[i] = f2bf(v);
            }
            *(u16x8*)(hread + hm * HREAD_PITCH + hk) = o;
        }
        __syncthreads();

        // ---- Phase B: acc = xproj_t + h_read @ Whh^T  (WMMA bf16) ----
        f32x8 acc = xp;
        if (t + 1 < T_DIM) xp = load_xp(t + 1);   // prefetch next step
        #pragma unroll
        for (int kf = 0; kf < 16; ++kf) {
            const unsigned short* ap =
                hread + lr * HREAD_PITCH + kf * 32 + hi * 8;
            u16x8 a0 = __builtin_bit_cast(u16x8, *(const u32x4*)(ap));
            u16x8 a1 = __builtin_bit_cast(u16x8, *(const u32x4*)(ap + 16));
            acc = wmma_bf16(cat16(a0, a1), bw[kf], acc);
        }

        // ---- Phase C: tanh + biases; write states + ring slot t&3 ----
        unsigned short* rdst = ring + s3 * SLOT_STRIDE;
        #pragma unroll
        for (int r = 0; r < 8; ++r) {
            const int ml = (hi ? r + 8 : r);          // 0..15 within tile
            const int m  = bOff + ml;                 // global batch
            const float h = fast_tanh(acc[r] + bsum);
            out_states[((size_t)m * T_DIM + t) * H_DIM + n] = h;
            rdst[ml * H_DIM + n] = f2bf(h);
            if (t == T_DIM - 1) out_last[(size_t)m * H_DIM + n] = h;
        }
        __syncthreads();
    }
}

// ---------------------------------------------------------------------------
extern "C" void kernel_launch(void* const* d_in, const int* in_sizes, int n_in,
                              void* d_out, int out_size, void* d_ws, size_t ws_size,
                              hipStream_t stream) {
    const float* x    = (const float*)d_in[0];
    const float* Wih  = (const float*)d_in[1];
    const float* Whh  = (const float*)d_in[2];
    const float* bih  = (const float*)d_in[3];
    const float* bhh  = (const float*)d_in[4];
    const float* taps = (const float*)d_in[5];

    unsigned short* WihB  = (unsigned short*)d_ws;           // 512 KB
    unsigned short* WhhB  = WihB + (size_t)H_DIM * I_DIM;    // 512 KB
    unsigned short* xproj = WhhB + (size_t)H_DIM * H_DIM;    // 32 MB (bf16)

    float* out_states = (float*)d_out;
    float* out_last   = out_states + (size_t)B_DIM * T_DIM * H_DIM;

    cvt_weights<<<(H_DIM * I_DIM + 255) / 256, 256, 0, stream>>>(
        Wih, Whh, WihB, WhhB);

    dim3 g1((B_DIM * T_DIM) / 64, H_DIM / 128);   // (512, 4)
    xproj_gemm<<<g1, 256, 0, stream>>>(x, WihB, xproj);

    // Actual LDS need is ~80.3 KB; request 161 KB so that two workgroups can
    // never co-reside on one 320 KB WGP -> each scan WG gets its own WGP.
    size_t smem_need = (size_t)(4 * SLOT_STRIDE + MB * HREAD_PITCH) *
                       sizeof(unsigned short);
    size_t smem = 161 * 1024;
    if (smem < smem_need) smem = smem_need;
    rnn_scan<<<B_DIM / MB, 1024, smem, stream>>>(WhhB, xproj, bih, bhh, taps,
                                                 out_states, out_last);
}
